// LSTMWithAttention_69578470195525
// MI455X (gfx1250) — matrix operand order repaired and checked by reading, not compiled, not run on hardware
//
#include <hip/hip_runtime.h>
#include <math.h>

// Problem constants (reference: B=8, S=2048, I=512, H=512)
#define BB   8
#define SS   2048
#define II   512
#define HH   512
#define G4   2048      // 4*H
#define MTOT 16384     // B*S
#define NWG_LSTM 32

typedef __attribute__((ext_vector_type(16))) __bf16 v16bf;
typedef __attribute__((ext_vector_type(8)))  float  v8f;
typedef __attribute__((ext_vector_type(4)))  unsigned int v4u;
typedef __attribute__((ext_vector_type(8)))  int    v8i;
typedef __attribute__((ext_vector_type(4)))  int    v4i;

// ---------------------------------------------------------------------------
// WMMA fragment helpers (wave32; K order below is this code's own, same for A and B)
// A 16x32 bf16: lane m(0-15)/m+16 -> row m; holds K = 16*(lane>>4) + 0..15 contiguous
// B 32x16 bf16: lane -> column n = lane&15; holds K = 16*(lane>>4) + 0..15 contiguous
// C/D 16x16 f32: VGPR i -> row i + 8*(lane>>4), col = lane&15
// ---------------------------------------------------------------------------
__device__ __forceinline__ v16bf load_frag_a(const __bf16* rowp, int kbase, int half) {
  const __bf16* p = rowp + kbase + 16 * half;
  v16bf a;
#pragma unroll
  for (int i = 0; i < 8; i++) a[i] = p[i];
#pragma unroll
  for (int i = 0; i < 8; i++) a[8 + i] = p[8 + i];
  return a;
}

__device__ __forceinline__ v16bf load_frag_b(const __bf16* colp, int koff) {
  const __bf16* p = colp + koff;
  v16bf b;
#pragma unroll
  for (int i = 0; i < 16; i++) b[i] = p[i];
  return b;
}

__device__ __forceinline__ v8f wmma_bf16(v16bf a, v16bf b, v8f c) {
  return __builtin_amdgcn_wmma_f32_16x16x32_bf16(false, a, false, b, (short)0, c,
                                                 false, false);
}

__device__ __forceinline__ float sigmoidf_(float x) {
  return 1.0f / (1.0f + expf(-x));
}

// ---------------------------------------------------------------------------
// Tensor Data Mover: 2D tile load Global -> LDS (CDNA5 ISA ch.8 D# layout).
// bf16 elements (data_size code 1 = 2 bytes). Issue from ONE wave only; waits
// on TENSORcnt are per-wave, so caller must s_wait_tensorcnt + __syncthreads.
//   lds_off  : byte offset of tile start within this WG's LDS allocation
//   gaddr    : global byte address of tile start
//   dim0     : tensor row length (elements)      [OOB bound, dim 0]
//   dim1     : tensor #rows                      [OOB bound, dim 1]
//   tile0/1  : tile size (elements / rows)
//   stride0  : elements between consecutive rows
// ---------------------------------------------------------------------------
__device__ __forceinline__ void tdm_load_2d_bf16(unsigned lds_off,
                                                 const void* gaddr,
                                                 unsigned dim0, unsigned dim1,
                                                 unsigned tile0, unsigned tile1,
                                                 unsigned stride0) {
  unsigned long long ga = (unsigned long long)gaddr;
  v4u g0;
  g0[0] = 1u;                                             // count=1, user D#
  g0[1] = lds_off;                                        // lds_addr
  g0[2] = (unsigned)(ga & 0xFFFFFFFFu);                   // global_addr lo
  g0[3] = (unsigned)((ga >> 32) & 0x01FFFFFFu) | (2u << 30);  // addr hi | type=2
  v8i g1;
  g1[0] = (int)(1u << 16);                                // wg_mask=0, data_size=2B
  g1[1] = (int)((dim0 & 0xFFFFu) << 16);                  // tensor_dim0 lo16
  g1[2] = (int)((dim0 >> 16) | ((dim1 & 0xFFFFu) << 16)); // dim0 hi / dim1 lo
  g1[3] = (int)((dim1 >> 16) | (tile0 << 16));            // dim1 hi / tile_dim0
  g1[4] = (int)(tile1 & 0xFFFFu);                         // tile_dim1, tile_dim2=0
  g1[5] = (int)stride0;                                   // tensor_dim0_stride lo
  g1[6] = 0;                                              // stride0 hi, stride1 lo
  g1[7] = 0;                                              // stride1 hi
  v4i gz4 = (v4i){0, 0, 0, 0};
  v8i gz8 = (v8i){0, 0, 0, 0, 0, 0, 0, 0};
  __builtin_amdgcn_tensor_load_to_lds(g0, g1, gz4, gz4, gz8, 0);
}

// ---------------------------------------------------------------------------
// Elementwise f32 -> bf16 conversion
// ---------------------------------------------------------------------------
__global__ void f32_to_bf16_kernel(const float* __restrict__ src,
                                   __bf16* __restrict__ dst, int n) {
  int i = blockIdx.x * blockDim.x + threadIdx.x;
  if (i < n) dst[i] = (__bf16)src[i];
}

// bias sum (b_ih + b_hh) and sync counter reset
__global__ void prep_kernel(const float* __restrict__ bih,
                            const float* __restrict__ bhh,
                            float* __restrict__ bsum, unsigned* sync, int n) {
  int i = blockIdx.x * blockDim.x + threadIdx.x;
  if (i < n) bsum[i] = bih[i] + bhh[i];
  if (i == 0) *sync = 0u;
}

// ---------------------------------------------------------------------------
// Generic bf16 GEMM (K == 512): out[m][n] = sum_k A[m][k]*Bw[n][k] (+ bias[n])
// Block = 256 thr (8 waves). B panel (64 rows x 512) staged to LDS via TDM.
// Wave tile 32x64 (2 M-subtiles x 4 N-subtiles = 8 wmma / K-step).
// Grid: (N/64, M/256). outF != null -> f32 output, else bf16 to outH.
// ---------------------------------------------------------------------------
__global__ __launch_bounds__(256) void gemm_bf16_kernel(
    const __bf16* __restrict__ A, const __bf16* __restrict__ Bw,
    const float* __restrict__ bias, float* __restrict__ outF,
    __bf16* __restrict__ outH, int M, int N, int K) {
  extern __shared__ char smem[];
  __bf16* Blds = (__bf16*)smem;  // [64][512] = 64KB, dynamic-LDS base offset 0

  const int lane = threadIdx.x & 31;
  const int wave = threadIdx.x >> 5;
  const int half = lane >> 4;
  const int nn = lane & 15;
  const int mbase = blockIdx.y * 256 + wave * 32;
  const int nbase = blockIdx.x * 64;

  // Stage B panel: rows [nbase, nbase+64) x K into LDS (single TDM op, wave 0)
  if (threadIdx.x < 32) {
    tdm_load_2d_bf16(/*lds_off=*/0u, Bw + (size_t)nbase * K,
                     (unsigned)K, (unsigned)N, (unsigned)K, 64u, (unsigned)K);
    __builtin_amdgcn_s_wait_tensorcnt(0);
  }
  __syncthreads();

  const __bf16* arow0 = A + (size_t)(mbase + nn) * K;
  const __bf16* arow1 = A + (size_t)(mbase + 16 + nn) * K;

  v8f acc[2][4];
#pragma unroll
  for (int s = 0; s < 2; s++)
#pragma unroll
    for (int j = 0; j < 4; j++)
      acc[s][j] = (v8f){0.f, 0.f, 0.f, 0.f, 0.f, 0.f, 0.f, 0.f};

  for (int k = 0; k < K; k += 32) {
    v16bf a0 = load_frag_a(arow0, k, half);
    v16bf a1 = load_frag_a(arow1, k, half);
#pragma unroll
    for (int j = 0; j < 4; j++) {
      v16bf b = load_frag_b(Blds + (size_t)(16 * j + nn) * K, k + 16 * half);
      acc[0][j] = wmma_bf16(a0, b, acc[0][j]);
      acc[1][j] = wmma_bf16(a1, b, acc[1][j]);
    }
  }

#pragma unroll
  for (int s = 0; s < 2; s++) {
#pragma unroll
    for (int j = 0; j < 4; j++) {
      int col = nbase + 16 * j + nn;
      float bv = bias ? bias[col] : 0.0f;
#pragma unroll
      for (int i = 0; i < 8; i++) {
        int row = mbase + 16 * s + i + 8 * half;
        float v = acc[s][j][i] + bv;
        if (outF) outF[(size_t)row * N + col] = v;
        else      outH[(size_t)row * N + col] = (__bf16)v;
      }
    }
  }
}

// ---------------------------------------------------------------------------
// Grid barrier (agent-scope atomic counter, release/acquire + threadfence)
// ---------------------------------------------------------------------------
__device__ __forceinline__ void grid_barrier(unsigned* cnt, unsigned target) {
  __threadfence();
  __syncthreads();
  if (threadIdx.x == 0) {
    __hip_atomic_fetch_add(cnt, 1u, __ATOMIC_RELEASE, __HIP_MEMORY_SCOPE_AGENT);
    while (__hip_atomic_load(cnt, __ATOMIC_ACQUIRE, __HIP_MEMORY_SCOPE_AGENT) <
           target) {
      __builtin_amdgcn_s_sleep(2);
    }
  }
  __syncthreads();
  __threadfence();
}

// ---------------------------------------------------------------------------
// Persistent LSTM kernel.
// 32 WGs x 128 threads (4 waves). WG w owns h columns [16w, 16w+16).
// Wave g computes gate-g tile [16(batch,padded) x 16 cols] via WMMA;
// W_hh slice (4 x 16 rows x 512) TDM-loaded to LDS. Cell state in registers.
// h shared globally via ping-pong bf16 buffers + grid barrier per step.
// ---------------------------------------------------------------------------
__global__ __launch_bounds__(128) void lstm_kernel(
    const __bf16* __restrict__ Whhbf, const float* __restrict__ xp,
    __bf16* __restrict__ hbuf /* 2 x [16][512] */,
    __bf16* __restrict__ lstmbf /* [B*S][H] */,
    __bf16* __restrict__ lstmbfT /* [B][H][S] */, unsigned* sync) {
  extern __shared__ char smem[];
  __bf16* wlds = (__bf16*)smem;                    // [64][512] = 64KB
  float* glds = (float*)(smem + 64 * 512 * 2);     // [4][16][16] = 4KB

  const int wg = blockIdx.x;
  const int tid = threadIdx.x;
  const int lane = tid & 31;
  const int wave = tid >> 5;  // == gate index
  const int half = lane >> 4;
  const int nn = lane & 15;

  // TDM preload of W_hh slice: 4 tiles (one per gate), 16 rows x 512 each.
  if (tid < 32) {
#pragma unroll
    for (int g = 0; g < 4; g++) {
      tdm_load_2d_bf16(/*lds_off=*/(unsigned)(g * 16 * 512 * 2),
                       Whhbf + ((size_t)(g * 512 + wg * 16)) * 512,
                       512u, (unsigned)G4, 512u, 16u, 512u);
    }
    __builtin_amdgcn_s_wait_tensorcnt(0);
  }
  // Zero-init both h ping-pong buffers for this WG's 16 columns, 16 padded rows
  for (int idx = tid; idx < 512; idx += 128) {
    int buf = idx >> 8, r = (idx >> 4) & 15, cc = wg * 16 + (idx & 15);
    hbuf[buf * (16 * 512) + r * 512 + cc] = (__bf16)0.0f;
  }
  grid_barrier(sync, NWG_LSTM * 1u);

  const int b = tid >> 4;          // batch for elementwise (tid < 128 -> b in 0..7)
  const int colL = tid & 15;
  const int colG = wg * 16 + colL;
  float cstate = 0.0f;

  for (int t = 0; t < SS; ++t) {
    const __bf16* hr = hbuf + (t & 1) * (16 * 512);
    __bf16* hw = hbuf + ((t + 1) & 1) * (16 * 512);

    // GEMM: gate tile [16x16] over K=512
    v8f acc = (v8f){0.f, 0.f, 0.f, 0.f, 0.f, 0.f, 0.f, 0.f};
    const __bf16* arow = hr + (size_t)nn * 512;
    const __bf16* brow = wlds + (size_t)(wave * 16 + nn) * 512;
    for (int k = 0; k < 512; k += 32) {
      v16bf a = load_frag_a(arow, k, half);
      v16bf bf = load_frag_b(brow, k + 16 * half);
      acc = wmma_bf16(a, bf, acc);
    }
#pragma unroll
    for (int i = 0; i < 8; i++)
      glds[wave * 256 + (i + 8 * half) * 16 + nn] = acc[i];
    __syncthreads();

    // Elementwise gate math for this WG's 8x16 block
    if (b < 8) {
      size_t xo = ((size_t)b * SS + t) * G4 + colG;
      float gi = glds[0 * 256 + b * 16 + colL] + xp[xo + 0 * 512];
      float gf = glds[1 * 256 + b * 16 + colL] + xp[xo + 1 * 512];
      float gg = glds[2 * 256 + b * 16 + colL] + xp[xo + 2 * 512];
      float go = glds[3 * 256 + b * 16 + colL] + xp[xo + 3 * 512];
      float iv = sigmoidf_(gi), fv = sigmoidf_(gf);
      float gv = tanhf(gg), ov = sigmoidf_(go);
      cstate = fv * cstate + iv * gv;
      float hv = ov * tanhf(cstate);
      __bf16 hb = (__bf16)hv;
      hw[b * 512 + colG] = hb;
      lstmbf[((size_t)b * SS + t) * HH + colG] = hb;
      lstmbfT[((size_t)b * HH + colG) * SS + t] = hb;
      // Prefetch next timestep's gate pre-activations (global_prefetch_b8)
      if (t + 1 < SS) {
        __builtin_prefetch(&xp[xo + G4], 0, 1);
        __builtin_prefetch(&xp[xo + G4 + 3 * 512], 0, 1);
      }
    }
    grid_barrier(sync, NWG_LSTM * (unsigned)(t + 2));
  }
}

// ---------------------------------------------------------------------------
// Attention kernel: one WG per (batch, 16-query tile). 256 thr = 8 waves.
// Pass 1: logits tiles (recomputable) -> row max.
// Pass 2: recompute, exp(l - max) -> LDS Pbf (bf16 [16][2048]), row sums.
// Pass 3: out = (Pbf @ lstm_outT) / rowsum  (f32 output).
// ---------------------------------------------------------------------------
__global__ __launch_bounds__(256) void attn_kernel(
    const __bf16* __restrict__ scoresbf /* [B*S][H] */,
    const __bf16* __restrict__ lstmbfT /* [B][H][S] */,
    float* __restrict__ out /* [B][S][H] */) {
  extern __shared__ char smem[];
  __bf16* Pbf = (__bf16*)smem;                      // [16][2048] = 64KB
  float* pmax = (float*)(smem + 16 * 2048 * 2);     // [16][16]
  float* rowmax = pmax + 256;                       // [16]
  float* rowsum = rowmax + 16;                      // [16]

  const int b = blockIdx.y;
  const int qt = blockIdx.x;
  const int tid = threadIdx.x;
  const int lane = tid & 31;
  const int wave = tid >> 5;
  const int half = lane >> 4;
  const int nn = lane & 15;
  const int qbase = qt * 16;

  const __bf16* Sb = scoresbf + (size_t)b * SS * HH;
  const __bf16* Vt = lstmbfT + (size_t)b * HH * SS;
  const __bf16* qrow = Sb + (size_t)(qbase + nn) * HH;

  for (int i = tid; i < 256 + 32; i += 256) {
    if (i < 256) pmax[i] = -3.0e38f;
    else if (i < 272) rowmax[i - 256] = -3.0e38f;
    else rowsum[i - 272] = 0.0f;
  }
  __syncthreads();

  // ---- pass 1: row maxima over all key tiles ----
  float lmax[8];
#pragma unroll
  for (int i = 0; i < 8; i++) lmax[i] = -3.0e38f;
  for (int kt = wave; kt < SS / 16; kt += 8) {
    v8f acc = (v8f){0.f, 0.f, 0.f, 0.f, 0.f, 0.f, 0.f, 0.f};
    const __bf16* krow = Sb + (size_t)(kt * 16 + nn) * HH;
    for (int k = 0; k < HH; k += 32) {
      v16bf a = load_frag_a(qrow, k, half);
      v16bf bf = load_frag_b(krow, k + 16 * half);
      acc = wmma_bf16(a, bf, acc);
    }
#pragma unroll
    for (int i = 0; i < 8; i++) lmax[i] = fmaxf(lmax[i], acc[i]);
  }
#pragma unroll
  for (int i = 0; i < 8; i++) {
    float* a = &pmax[(i + 8 * half) * 16 + nn];
    float v = lmax[i];
    float old = *a;
    while (v > old) {
      if (__hip_atomic_compare_exchange_strong(a, &old, v, __ATOMIC_RELAXED,
                                               __ATOMIC_RELAXED,
                                               __HIP_MEMORY_SCOPE_WORKGROUP))
        break;
    }
  }
  __syncthreads();
  if (tid < 16) {
    float m = -3.0e38f;
    for (int j = 0; j < 16; j++) m = fmaxf(m, pmax[tid * 16 + j]);
    rowmax[tid] = m;
  }
  __syncthreads();

  // ---- pass 2: recompute logits, exponentiate, store probabilities ----
  float lsum[8];
#pragma unroll
  for (int i = 0; i < 8; i++) lsum[i] = 0.0f;
  for (int kt = wave; kt < SS / 16; kt += 8) {
    v8f acc = (v8f){0.f, 0.f, 0.f, 0.f, 0.f, 0.f, 0.f, 0.f};
    const __bf16* krow = Sb + (size_t)(kt * 16 + nn) * HH;
    for (int k = 0; k < HH; k += 32) {
      v16bf a = load_frag_a(qrow, k, half);
      v16bf bf = load_frag_b(krow, k + 16 * half);
      acc = wmma_bf16(a, bf, acc);
    }
#pragma unroll
    for (int i = 0; i < 8; i++) {
      int row = i + 8 * half;
      float p = expf(acc[i] - rowmax[row]);
      lsum[i] += p;
      Pbf[row * SS + kt * 16 + nn] = (__bf16)p;
    }
  }
#pragma unroll
  for (int i = 0; i < 8; i++) atomicAdd(&rowsum[i + 8 * half], lsum[i]);
  __syncthreads();

  // ---- pass 3: out tile = P @ V, scaled by 1/rowsum ----
  const __bf16* prow = Pbf + (size_t)nn * SS;
  for (int nt = wave; nt < HH / 16; nt += 8) {
    int hbase = nt * 16;
    v8f acc = (v8f){0.f, 0.f, 0.f, 0.f, 0.f, 0.f, 0.f, 0.f};
    const __bf16* vrow = Vt + (size_t)(hbase + nn) * SS;
    for (int k = 0; k < SS; k += 32) {
      v16bf a = load_frag_a(prow, k, half);
      v16bf bf = load_frag_b(vrow, k + 16 * half);
      acc = wmma_bf16(a, bf, acc);
    }
#pragma unroll
    for (int i = 0; i < 8; i++) {
      int row = i + 8 * half;
      out[((size_t)b * SS + qbase + row) * HH + hbase + nn] =
          acc[i] / rowsum[row];
    }
  }
}

// ---------------------------------------------------------------------------
// Host-side launcher
// ---------------------------------------------------------------------------
extern "C" void kernel_launch(void* const* d_in, const int* in_sizes, int n_in,
                              void* d_out, int out_size, void* d_ws,
                              size_t ws_size, hipStream_t stream) {
  (void)in_sizes; (void)n_in; (void)out_size;
  const float* X   = (const float*)d_in[0];
  const float* Wih = (const float*)d_in[1];
  const float* Whh = (const float*)d_in[2];
  const float* bih = (const float*)d_in[3];
  const float* bhh = (const float*)d_in[4];
  const float* Wa  = (const float*)d_in[5];
  const float* ba  = (const float*)d_in[6];
  float* out = (float*)d_out;

  char* ws = (char*)d_ws;
  size_t off = 0;
  auto alloc = [&](size_t bytes) -> char* {
    char* p = ws + off;
    off += (bytes + 255) & ~(size_t)255;
    return p;
  };
  __bf16* Xbf      = (__bf16*)alloc((size_t)MTOT * II * 2);
  __bf16* Wihbf    = (__bf16*)alloc((size_t)G4 * II * 2);
  __bf16* Whhbf    = (__bf16*)alloc((size_t)G4 * HH * 2);
  __bf16* Wabf     = (__bf16*)alloc((size_t)HH * HH * 2);
  float*  bsum     = (float*)alloc((size_t)G4 * 4);
  __bf16* hbuf     = (__bf16*)alloc((size_t)2 * 16 * 512 * 2);
  unsigned* sync   = (unsigned*)alloc(256);
  float*  xp       = (float*)alloc((size_t)MTOT * G4 * 4);
  __bf16* lstmbf   = (__bf16*)alloc((size_t)MTOT * HH * 2);
  __bf16* lstmbfT  = (__bf16*)alloc((size_t)BB * HH * SS * 2);
  __bf16* scoresbf = (__bf16*)alloc((size_t)MTOT * HH * 2);
  if (off > ws_size) return;  // insufficient scratch; bail deterministically

  const size_t gemm_smem = 64 * 512 * 2;                                // 65536
  const size_t lstm_smem = 64 * 512 * 2 + 4 * 16 * 16 * 4;              // 69632
  const size_t attn_smem = 16 * 2048 * 2 + (256 + 16 + 16) * 4;         // 66688
  (void)hipFuncSetAttribute((const void*)gemm_bf16_kernel,
                            hipFuncAttributeMaxDynamicSharedMemorySize,
                            (int)gemm_smem);
  (void)hipFuncSetAttribute((const void*)lstm_kernel,
                            hipFuncAttributeMaxDynamicSharedMemorySize,
                            (int)lstm_smem);
  (void)hipFuncSetAttribute((const void*)attn_kernel,
                            hipFuncAttributeMaxDynamicSharedMemorySize,
                            (int)attn_smem);

  // 1) conversions + bias prep
  int nX = MTOT * II;
  f32_to_bf16_kernel<<<(nX + 255) / 256, 256, 0, stream>>>(X, Xbf, nX);
  f32_to_bf16_kernel<<<(G4 * II + 255) / 256, 256, 0, stream>>>(Wih, Wihbf,
                                                                G4 * II);
  f32_to_bf16_kernel<<<(G4 * HH + 255) / 256, 256, 0, stream>>>(Whh, Whhbf,
                                                                G4 * HH);
  f32_to_bf16_kernel<<<(HH * HH + 255) / 256, 256, 0, stream>>>(Wa, Wabf,
                                                                HH * HH);
  prep_kernel<<<(G4 + 255) / 256, 256, 0, stream>>>(bih, bhh, bsum, sync, G4);

  // 2) xp = X @ W_ih^T + (b_ih + b_hh)   [f32, 16384 x 2048]
  gemm_bf16_kernel<<<dim3(G4 / 64, MTOT / 256), 256, gemm_smem, stream>>>(
      Xbf, Wihbf, bsum, xp, (/*unused*/ __bf16*)nullptr, MTOT, G4, II);

  // 3) LSTM recurrence (persistent, grid-synced)
  lstm_kernel<<<NWG_LSTM, 128, lstm_smem, stream>>>(Whhbf, xp, hbuf, lstmbf,
                                                    lstmbfT, sync);

  // 4) scores = lstm_out @ Wa^T + ba  -> bf16 [16384 x 512]
  gemm_bf16_kernel<<<dim3(HH / 64, MTOT / 256), 256, gemm_smem, stream>>>(
      lstmbf, Wabf, ba, (float*)nullptr, scoresbf, MTOT, HH, HH);

  // 5) attention: softmax(scores scores^T) @ lstm_out -> d_out (f32)
  attn_kernel<<<dim3(SS / 16, BB), 256, attn_smem, stream>>>(scoresbf, lstmbfT,
                                                             out);
}
